// RingAttention_52132313039110
// MI455X (gfx1250) — matrix operand order repaired and checked by reading, not compiled, not run on hardware
//
#include <hip/hip_runtime.h>

typedef __attribute__((ext_vector_type(16))) _Float16 v16h;
typedef __attribute__((ext_vector_type(8)))  float    v8f;

#define HEADS 16
#define DHEAD 64
#define BATCH 2
#define SEQ   2048
#define DIM   1024
#define DIM3  3072
#define MTOT  (BATCH * SEQ)   // 4096

// ---------------------------------------------------------------------------
// CDNA5 async global->LDS copy helpers (ASYNCcnt path, bypasses VGPRs)
// ---------------------------------------------------------------------------
__device__ __forceinline__ void async_copy32(void* ldsp, const void* g) {
    unsigned l = (unsigned)(uintptr_t)ldsp;                 // low 32 bits = LDS offset
    unsigned long long ga = (unsigned long long)(uintptr_t)g;
    asm volatile("global_load_async_to_lds_b128 %0, %1, off"
                 :: "v"(l), "v"(ga) : "memory");
    asm volatile("global_load_async_to_lds_b128 %0, %1, off offset:16"
                 :: "v"(l), "v"(ga) : "memory");
}
__device__ __forceinline__ void wait_async0() {
    asm volatile("s_wait_asynccnt 0x0" ::: "memory");
}

// ---------------------------------------------------------------------------
// 1) RMSNorm: xn[r][c] = x[r][c] / max(||x_r||,eps) * sqrt(DIM) * gamma[c]
// ---------------------------------------------------------------------------
__global__ __launch_bounds__(256) void norm_kernel(const float* __restrict__ x,
                                                   const float* __restrict__ gamma,
                                                   _Float16* __restrict__ xn) {
    __shared__ float red[256];
    const int row = blockIdx.x;
    const float* xr = x + (size_t)row * DIM;
    float ss = 0.f;
    for (int c = threadIdx.x; c < DIM; c += 256) { float v = xr[c]; ss += v * v; }
    red[threadIdx.x] = ss;
    __syncthreads();
    for (int s = 128; s > 0; s >>= 1) {
        if (threadIdx.x < (unsigned)s) red[threadIdx.x] += red[threadIdx.x + s];
        __syncthreads();
    }
    const float scale = 32.0f / fmaxf(sqrtf(red[0]), 1e-12f);  // sqrt(1024)=32
    for (int c = threadIdx.x; c < DIM; c += 256)
        xn[(size_t)row * DIM + c] = (_Float16)(xr[c] * scale * gamma[c]);
}

// ---------------------------------------------------------------------------
// 2) Weight convert f32 -> f16 with transpose: wt[n][k] = w[k][n]
// ---------------------------------------------------------------------------
__global__ __launch_bounds__(256) void transpose_f32_to_f16(const float* __restrict__ w,
                                                            _Float16* __restrict__ wt,
                                                            int K, int N) {
    int idx = blockIdx.x * 256 + threadIdx.x;
    if (idx >= K * N) return;
    int k = idx / N, n = idx % N;
    wt[(size_t)n * K + k] = (_Float16)w[idx];
}

// ---------------------------------------------------------------------------
// 3) QKV GEMM: [4096 x 1024] x [1024 x 3072] via WMMA f32_16x16x32_f16.
//    Block 128x128, 8 waves, each wave 64x32 (4x2 WMMA tiles).
//    Async double-buffered LDS staging; epilogue scatters q/k/v-transposed.
// ---------------------------------------------------------------------------
__global__ __launch_bounds__(256) void qkv_gemm_kernel(const _Float16* __restrict__ A,
                                                       const _Float16* __restrict__ Bt,
                                                       _Float16* __restrict__ qb,
                                                       _Float16* __restrict__ kb,
                                                       _Float16* __restrict__ vtb) {
    __shared__ _Float16 lA[2][128 * 32];
    __shared__ _Float16 lB[2][128 * 32];
    const int bm = blockIdx.y * 128;
    const int bn = blockIdx.x * 128;
    const int tid = threadIdx.x;
    const int wid = tid >> 5;
    const int lane = tid & 31;
    const int wm = wid >> 2;      // 0..1  (64 rows each)
    const int wn = wid & 3;       // 0..3  (32 cols each)
    const int lhalf = lane >> 4;  // half-wave group
    const int l16 = lane & 15;

    v8f acc[4][2];
#pragma unroll
    for (int i = 0; i < 4; i++)
#pragma unroll
        for (int j = 0; j < 2; j++)
#pragma unroll
            for (int e = 0; e < 8; e++) acc[i][j][e] = 0.f;

    const int sr = tid >> 1;         // staging row 0..127
    const int sc = (tid & 1) * 16;   // staging col 0 / 16
    const _Float16* Abase = A + (size_t)(bm + sr) * DIM + sc;
    const _Float16* Bbase = Bt + (size_t)(bn + sr) * DIM + sc;

    // prologue: async-load tile 0
    async_copy32(&lA[0][sr * 32 + sc], Abase);
    async_copy32(&lB[0][sr * 32 + sc], Bbase);

    int buf = 0;
    for (int k0 = 0; k0 < DIM; k0 += 32, buf ^= 1) {
        wait_async0();
        __syncthreads();
        if (k0 + 32 < DIM) {   // prefetch next tile into the other buffer
            async_copy32(&lA[buf ^ 1][sr * 32 + sc], Abase + k0 + 32);
            async_copy32(&lB[buf ^ 1][sr * 32 + sc], Bbase + k0 + 32);
        }

        v16h bf[2];
#pragma unroll
        for (int j = 0; j < 2; j++) {
            int n = wn * 32 + j * 16 + l16;
            *(uint4*)&bf[j]       = *(const uint4*)&lB[buf][n * 32 + 8 * lhalf];
            *((uint4*)&bf[j] + 1) = *(const uint4*)&lB[buf][n * 32 + 16 + 8 * lhalf];
        }
#pragma unroll
        for (int i = 0; i < 4; i++) {
            int r = wm * 64 + i * 16 + l16;
            v16h af;
            *(uint4*)&af       = *(const uint4*)&lA[buf][r * 32 + 8 * lhalf];
            *((uint4*)&af + 1) = *(const uint4*)&lA[buf][r * 32 + 16 + 8 * lhalf];
            acc[i][0] = __builtin_amdgcn_wmma_f32_16x16x32_f16(false, af, false, bf[0],
                                                               (short)0, acc[i][0], false, false);
            acc[i][1] = __builtin_amdgcn_wmma_f32_16x16x32_f16(false, af, false, bf[1],
                                                               (short)0, acc[i][1], false, false);
        }
    }

    const float qscale = 0.125f;  // 1/sqrt(64)
#pragma unroll
    for (int i = 0; i < 4; i++)
#pragma unroll
        for (int j = 0; j < 2; j++) {
            int col = bn + wn * 32 + j * 16 + l16;
            int which = col >> 10;     // 0=q 1=k 2=v
            int within = col & 1023;
            int h = within >> 6;
            int d = within & 63;
#pragma unroll
            for (int e = 0; e < 8; e++) {
                int m = bm + wm * 64 + i * 16 + e + 8 * lhalf;
                int b = m >> 11;
                int t = m & 2047;
                float val = acc[i][j][e];
                size_t bh = (size_t)(b * HEADS + h);
                if (which == 0)
                    qb[(bh * SEQ + t) * DHEAD + d] = (_Float16)(val * qscale);
                else if (which == 1)
                    kb[(bh * SEQ + t) * DHEAD + d] = (_Float16)val;
                else
                    vtb[(bh * DHEAD + d) * SEQ + t] = (_Float16)val;
            }
        }
}

// ---------------------------------------------------------------------------
// 4) Causal flash attention. One workgroup = (b,h) x 64-row Q tile; 4 waves,
//    each wave owns 16 query rows. Key loop in 32-column tiles; K/V staged
//    via async global->LDS.
// ---------------------------------------------------------------------------
__global__ __launch_bounds__(128) void attn_kernel(const _Float16* __restrict__ qb,
                                                   const _Float16* __restrict__ kb,
                                                   const _Float16* __restrict__ vtb,
                                                   _Float16* __restrict__ ao) {
    __shared__ _Float16 lQ[64 * 64];
    __shared__ _Float16 lK[32 * 64];
    __shared__ _Float16 lV[64 * 32];       // V transposed: [d][key]
    __shared__ _Float16 lP[4][16 * 32];    // per-wave P staging

    const int bh = blockIdx.x;
    const int b = bh >> 4;
    const int h = bh & 15;
    const int qt = blockIdx.y;
    const int qs = qt * 64;
    const int tid = threadIdx.x;
    const int wid = tid >> 5;
    const int lane = tid & 31;
    const int lhalf = lane >> 4;
    const int l16 = lane & 15;

    const _Float16* qp = qb + (size_t)bh * SEQ * DHEAD;
    const _Float16* kp = kb + (size_t)bh * SEQ * DHEAD;
    const _Float16* vp = vtb + (size_t)bh * DHEAD * SEQ;

    {   // Q tile 64x64
        int r = tid >> 1;
        int c = (tid & 1) * 32;
        const _Float16* src = qp + (size_t)(qs + r) * DHEAD + c;
        *(uint4*)&lQ[r * 64 + c]      = *(const uint4*)(src);
        *(uint4*)&lQ[r * 64 + c + 8]  = *(const uint4*)(src + 8);
        *(uint4*)&lQ[r * 64 + c + 16] = *(const uint4*)(src + 16);
        *(uint4*)&lQ[r * 64 + c + 24] = *(const uint4*)(src + 24);
    }

    float mrow[8], lrow[8];
    v8f oacc[4];
#pragma unroll
    for (int v = 0; v < 8; v++) { mrow[v] = -1e30f; lrow[v] = 0.f; }
#pragma unroll
    for (int t = 0; t < 4; t++)
#pragma unroll
        for (int e = 0; e < 8; e++) oacc[t][e] = 0.f;

    // per-thread staging coordinates
    const int kr = tid >> 2, kc = (tid & 3) * 16;   // K tile 32x64
    const int vr = tid >> 1, vc = (tid & 1) * 16;   // Vt tile 64x32

    const int ktmax = 2 * qt + 1;
    for (int kt = 0; kt <= ktmax; kt++) {
        __syncthreads();   // previous tiles fully consumed
        async_copy32(&lK[kr * 64 + kc], kp + (size_t)(kt * 32 + kr) * DHEAD + kc);
        async_copy32(&lV[vr * 32 + vc], vp + (size_t)vr * SEQ + kt * 32 + vc);
        wait_async0();
        __syncthreads();

        // S = Q_w (16x64) . K^T  -> two 16x16 f32 tiles
        v8f s0, s1;
#pragma unroll
        for (int e = 0; e < 8; e++) { s0[e] = 0.f; s1[e] = 0.f; }
#pragma unroll
        for (int kk = 0; kk < 64; kk += 32) {
            int r = wid * 16 + l16;
            v16h af, bf0, bf1;
            *(uint4*)&af        = *(const uint4*)&lQ[r * 64 + kk + 8 * lhalf];
            *((uint4*)&af + 1)  = *(const uint4*)&lQ[r * 64 + kk + 16 + 8 * lhalf];
            *(uint4*)&bf0       = *(const uint4*)&lK[l16 * 64 + kk + 8 * lhalf];
            *((uint4*)&bf0 + 1) = *(const uint4*)&lK[l16 * 64 + kk + 16 + 8 * lhalf];
            *(uint4*)&bf1       = *(const uint4*)&lK[(16 + l16) * 64 + kk + 8 * lhalf];
            *((uint4*)&bf1 + 1) = *(const uint4*)&lK[(16 + l16) * 64 + kk + 16 + 8 * lhalf];
            s0 = __builtin_amdgcn_wmma_f32_16x16x32_f16(false, af, false, bf0, (short)0, s0, false, false);
            s1 = __builtin_amdgcn_wmma_f32_16x16x32_f16(false, af, false, bf1, (short)0, s1, false, false);
        }

        // causal mask + online softmax (rows = v + 8*lhalf, cols across 16 lanes)
        const int kbase = kt * 32;
        float pnew0[8], pnew1[8];
#pragma unroll
        for (int v = 0; v < 8; v++) {
            int qrow = qs + wid * 16 + v + 8 * lhalf;
            float a0 = s0[v], a1 = s1[v];
            if (kbase + l16 > qrow)      a0 = -1e30f;
            if (kbase + 16 + l16 > qrow) a1 = -1e30f;
            float mx = fmaxf(a0, a1);
#pragma unroll
            for (int off = 8; off > 0; off >>= 1) mx = fmaxf(mx, __shfl_xor(mx, off, 16));
            float mnew = fmaxf(mrow[v], mx);
            float corr = __expf(mrow[v] - mnew);
            float p0 = __expf(a0 - mnew);
            float p1 = __expf(a1 - mnew);
            float rs = p0 + p1;
#pragma unroll
            for (int off = 8; off > 0; off >>= 1) rs += __shfl_xor(rs, off, 16);
            lrow[v] = lrow[v] * corr + rs;
            mrow[v] = mnew;
#pragma unroll
            for (int t = 0; t < 4; t++) oacc[t][v] *= corr;
            pnew0[v] = p0; pnew1[v] = p1;
        }

        // stage P (per-wave LDS), then O += P x V
        _Float16* P = &lP[wid][0];
#pragma unroll
        for (int v = 0; v < 8; v++) {
            P[(v + 8 * lhalf) * 32 + l16]      = (_Float16)pnew0[v];
            P[(v + 8 * lhalf) * 32 + 16 + l16] = (_Float16)pnew1[v];
        }
        asm volatile("s_wait_dscnt 0x0" ::: "memory");
        v16h pf;
        *(uint4*)&pf       = *(const uint4*)&P[l16 * 32 + 8 * lhalf];
        *((uint4*)&pf + 1) = *(const uint4*)&P[l16 * 32 + 16 + 8 * lhalf];
#pragma unroll
        for (int t = 0; t < 4; t++) {
            int d = t * 16 + l16;
            v16h vf;
            *(uint4*)&vf       = *(const uint4*)&lV[d * 32 + 8 * lhalf];
            *((uint4*)&vf + 1) = *(const uint4*)&lV[d * 32 + 16 + 8 * lhalf];
            oacc[t] = __builtin_amdgcn_wmma_f32_16x16x32_f16(false, pf, false, vf,
                                                             (short)0, oacc[t], false, false);
        }
    }

    // finalize: O /= l, write to ao[token][h*64+d] (f16)
#pragma unroll
    for (int v = 0; v < 8; v++) {
        int token = b * SEQ + qs + wid * 16 + v + 8 * lhalf;
        float inv = 1.0f / lrow[v];
#pragma unroll
        for (int t = 0; t < 4; t++) {
            int col = h * DHEAD + t * 16 + l16;
            ao[(size_t)token * DIM + col] = (_Float16)(oacc[t][v] * inv);
        }
    }
}

// ---------------------------------------------------------------------------
// 5) Output GEMM: [4096 x 1024] x [1024 x 1024] -> fp32 out
// ---------------------------------------------------------------------------
__global__ __launch_bounds__(256) void out_gemm_kernel(const _Float16* __restrict__ A,
                                                       const _Float16* __restrict__ Bt,
                                                       float* __restrict__ out) {
    __shared__ _Float16 lA[2][128 * 32];
    __shared__ _Float16 lB[2][128 * 32];
    const int bm = blockIdx.y * 128;
    const int bn = blockIdx.x * 128;
    const int tid = threadIdx.x;
    const int wid = tid >> 5;
    const int lane = tid & 31;
    const int wm = wid >> 2;
    const int wn = wid & 3;
    const int lhalf = lane >> 4;
    const int l16 = lane & 15;

    v8f acc[4][2];
#pragma unroll
    for (int i = 0; i < 4; i++)
#pragma unroll
        for (int j = 0; j < 2; j++)
#pragma unroll
            for (int e = 0; e < 8; e++) acc[i][j][e] = 0.f;

    const int sr = tid >> 1;
    const int sc = (tid & 1) * 16;
    const _Float16* Abase = A + (size_t)(bm + sr) * DIM + sc;
    const _Float16* Bbase = Bt + (size_t)(bn + sr) * DIM + sc;

    async_copy32(&lA[0][sr * 32 + sc], Abase);
    async_copy32(&lB[0][sr * 32 + sc], Bbase);

    int buf = 0;
    for (int k0 = 0; k0 < DIM; k0 += 32, buf ^= 1) {
        wait_async0();
        __syncthreads();
        if (k0 + 32 < DIM) {
            async_copy32(&lA[buf ^ 1][sr * 32 + sc], Abase + k0 + 32);
            async_copy32(&lB[buf ^ 1][sr * 32 + sc], Bbase + k0 + 32);
        }

        v16h bf[2];
#pragma unroll
        for (int j = 0; j < 2; j++) {
            int n = wn * 32 + j * 16 + l16;
            *(uint4*)&bf[j]       = *(const uint4*)&lB[buf][n * 32 + 8 * lhalf];
            *((uint4*)&bf[j] + 1) = *(const uint4*)&lB[buf][n * 32 + 16 + 8 * lhalf];
        }
#pragma unroll
        for (int i = 0; i < 4; i++) {
            int r = wm * 64 + i * 16 + l16;
            v16h af;
            *(uint4*)&af       = *(const uint4*)&lA[buf][r * 32 + 8 * lhalf];
            *((uint4*)&af + 1) = *(const uint4*)&lA[buf][r * 32 + 16 + 8 * lhalf];
            acc[i][0] = __builtin_amdgcn_wmma_f32_16x16x32_f16(false, af, false, bf[0],
                                                               (short)0, acc[i][0], false, false);
            acc[i][1] = __builtin_amdgcn_wmma_f32_16x16x32_f16(false, af, false, bf[1],
                                                               (short)0, acc[i][1], false, false);
        }
    }

#pragma unroll
    for (int i = 0; i < 4; i++)
#pragma unroll
        for (int j = 0; j < 2; j++) {
            int col = bn + wn * 32 + j * 16 + l16;
#pragma unroll
            for (int e = 0; e < 8; e++) {
                int m = bm + wm * 64 + i * 16 + e + 8 * lhalf;
                out[(size_t)m * DIM + col] = acc[i][j][e];
            }
        }
}

// ---------------------------------------------------------------------------
// Launcher
// ---------------------------------------------------------------------------
extern "C" void kernel_launch(void* const* d_in, const int* in_sizes, int n_in,
                              void* d_out, int out_size, void* d_ws, size_t ws_size,
                              hipStream_t stream) {
    const float* x     = (const float*)d_in[0];
    const float* gamma = (const float*)d_in[1];
    const float* w_qkv = (const float*)d_in[2];
    const float* w_out = (const float*)d_in[3];
    float* out = (float*)d_out;

    char* ws = (char*)d_ws;
    _Float16* xn    = (_Float16*)(ws);                         // 8 MB
    _Float16* wqkvt = (_Float16*)(ws + ((size_t)8  << 20));    // 6 MB
    _Float16* woutt = (_Float16*)(ws + ((size_t)14 << 20));    // 2 MB
    _Float16* qbuf  = (_Float16*)(ws + ((size_t)16 << 20));    // 8 MB
    _Float16* kbuf  = (_Float16*)(ws + ((size_t)24 << 20));    // 8 MB
    _Float16* vtbuf = (_Float16*)(ws + ((size_t)32 << 20));    // 8 MB
    _Float16* aobuf = (_Float16*)(ws + ((size_t)40 << 20));    // 8 MB

    norm_kernel<<<MTOT, 256, 0, stream>>>(x, gamma, xn);
    transpose_f32_to_f16<<<(DIM * DIM3 + 255) / 256, 256, 0, stream>>>(w_qkv, wqkvt, DIM, DIM3);
    transpose_f32_to_f16<<<(DIM * DIM + 255) / 256, 256, 0, stream>>>(w_out, woutt, DIM, DIM);
    qkv_gemm_kernel<<<dim3(DIM3 / 128, MTOT / 128), 256, 0, stream>>>(xn, wqkvt, qbuf, kbuf, vtbuf);
    attn_kernel<<<dim3(BATCH * HEADS, SEQ / 64), 128, 0, stream>>>(qbuf, kbuf, vtbuf, aobuf);
    out_gemm_kernel<<<dim3(DIM / 128, MTOT / 128), 256, 0, stream>>>(aobuf, woutt, out);
}
